// OrbitCollapse_63273458205272
// MI455X (gfx1250) — matrix operand order repaired and checked by reading, not compiled, not run on hardware
//
#include <hip/hip_runtime.h>
#include <hip/hip_bf16.h>
#include <math.h>

// ---------------------------------------------------------------------------
// Same-class mean pairwise Euclidean distance.  N=8192, D=256, 64 classes.
// Dominant cost: Gram = E @ E^T (34.4 GFLOP).  E is 8 MB -> L2-resident on
// MI455X (192 MB L2) => compute-bound, so pick the matrix pipe the chip
// rewards:
//   * primary path: split-bf16 (x = hi + lo, dot ~= hi*hi + hi*lo + lo*hi,
//     fp32 accumulate) on V_WMMA_F32_16X16X32_BF16 -> 3 WMMAs per K=32
//     instead of 8 V_WMMA_F32_16X16X4_F32 per K=32 (~2.7x fewer matrix ops
//     on the ~8x-deeper op), with ~16+ mantissa bits of effective precision.
//   * fallback path (if ws too small for the bf16 copies): native fp32 WMMA.
// Round-2 fix: register double-buffer across the WMMA section spilled to
// scratch -> removed.  Round-3 fix: the direct copy serialized as
// load/wait(0)/store per 16B; stage the 8 quads in distinct temps *within*
// the copy phase only, so loads issue back-to-back and drain with partial
// s_wait_loadcnt (no long live ranges, no spill).
// ---------------------------------------------------------------------------

typedef float  v2f   __attribute__((ext_vector_type(2)));
typedef float  v8f   __attribute__((ext_vector_type(8)));
typedef __bf16 v16bf __attribute__((ext_vector_type(16)));

#define NROW 8192
#define DIM  256
#define BT   128    // block tile (rows == cols)
#define KC   32     // K chunk staged in LDS (fp32 path)
#define LDSS 36     // fp32 LDS row stride (floats): 16B-aligned, conflict-free
#define LBS  40     // bf16 LDS row stride (ushorts = 80B): 16B-aligned, conflict-free

// ws layout (bytes):
//   [0,8)            : sum, count (float)
//   [16, 16+32768)   : x2[8192] (float)
//   [32784, +4MB)    : Ehi  (bf16 as ushort)   -- bf16 path only
//   [.. , +4MB)      : Elo  (bf16 as ushort)   -- bf16 path only

__global__ void oc_init_kernel(float* ws) {
    ws[0] = 0.0f;
    ws[1] = 0.0f;
}

// One wave per row: exact fp32 row norms.
__global__ __launch_bounds__(256) void oc_rowsq_kernel(const float* __restrict__ E,
                                                       float* __restrict__ x2) {
    const int wave = threadIdx.x >> 5;
    const int lane = threadIdx.x & 31;
    const int row  = blockIdx.x * 8 + wave;
    const float* p = E + (size_t)row * DIM;
    float s = 0.0f;
    #pragma unroll
    for (int k = lane; k < DIM; k += 32) {
        float v = p[k];
        s = fmaf(v, v, s);
    }
    #pragma unroll
    for (int off = 16; off > 0; off >>= 1)
        s += __shfl_down(s, off, 32);
    if (lane == 0) x2[row] = s;
}

static __device__ __forceinline__ unsigned short f32_to_bf16_rne(float x) {
    unsigned u = __float_as_uint(x);
    u += 0x7FFFu + ((u >> 16) & 1u);      // round-to-nearest-even
    return (unsigned short)(u >> 16);
}

// Split E into bf16 hi + bf16 lo (lo = x - float(hi), RNE both times).
__global__ __launch_bounds__(256) void oc_split_kernel(const float* __restrict__ E,
                                                       unsigned short* __restrict__ hi,
                                                       unsigned short* __restrict__ lo) {
    const int i4 = blockIdx.x * 256 + threadIdx.x;   // float4 index
    const float4 v = ((const float4*)E)[i4];
    const float f[4] = {v.x, v.y, v.z, v.w};
    unsigned short hs[4], ls[4];
    #pragma unroll
    for (int k = 0; k < 4; ++k) {
        const unsigned short hb = f32_to_bf16_rne(f[k]);
        const float hf = __uint_as_float((unsigned)hb << 16);
        hs[k] = hb;
        ls[k] = f32_to_bf16_rne(f[k] - hf);
    }
    ((ushort4*)hi)[i4] = make_ushort4(hs[0], hs[1], hs[2], hs[3]);
    ((ushort4*)lo)[i4] = make_ushort4(ls[0], ls[1], ls[2], ls[3]);
}

// ---------------------------------------------------------------------------
// Primary path: split-bf16 Gram tile, V_WMMA_F32_16X16X32_BF16.
// 128x128 tile per workgroup; 8 waves, each owns a 32x64 subtile
// (2x4 grid of 16x16 accumulators).  Upper-triangular block tiles only.
// ---------------------------------------------------------------------------
__global__ __launch_bounds__(256) void oc_pairdist_bf16_kernel(
        const unsigned short* __restrict__ Ehi,
        const unsigned short* __restrict__ Elo,
        const int*   __restrict__ labels,
        const float* __restrict__ x2,
        float*       __restrict__ accum) {
    const int bx = blockIdx.x;
    const int by = blockIdx.y;
    if (bx < by) return;

    __shared__ __align__(16) unsigned short ldsAhi[BT * LBS];
    __shared__ __align__(16) unsigned short ldsAlo[BT * LBS];
    __shared__ __align__(16) unsigned short ldsBhi[BT * LBS];
    __shared__ __align__(16) unsigned short ldsBlo[BT * LBS];
    __shared__ float sx2A[BT], sx2B[BT];
    __shared__ int   slA[BT],  slB[BT];
    __shared__ float red_s[8];
    __shared__ int   red_c[8];

    const int tid  = threadIdx.x;
    const int wave = tid >> 5;
    const int lane = tid & 31;
    const int half = lane >> 4;
    const int l16  = lane & 15;
    const int wr   = wave >> 1;   // 0..3 : 32-row band
    const int wc   = wave & 1;    // 0..1 : 64-col band

    if (tid < BT) {
        sx2A[tid] = x2[by * BT + tid];
        slA[tid]  = labels[by * BT + tid];
    } else {
        const int t = tid - BT;
        sx2B[t] = x2[bx * BT + t];
        slB[t]  = labels[bx * BT + t];
    }

    // Each thread copies 2 b128 per buffer per chunk (row = fi>>2, c16 = fi&3).
    const int r0 = tid >> 2,         c0 = (tid & 3) * 8;
    const int r1 = (tid + 256) >> 2, c1 = ((tid + 256) & 3) * 8;
    const size_t rowA0 = (size_t)(by * BT + r0) * DIM + c0;
    const size_t rowA1 = (size_t)(by * BT + r1) * DIM + c1;
    const size_t rowB0 = (size_t)(bx * BT + r0) * DIM + c0;
    const size_t rowB1 = (size_t)(bx * BT + r1) * DIM + c1;
    const int ld0 = r0 * LBS + c0;
    const int ld1 = r1 * LBS + c1;

    union Frag { uint4 q[2]; v16bf v; };
    // ISA 16-bit A 16x32: lane holds K[8*half..+8) in v0-3, K[16+8*half..+8) in v4-7.
    auto loadA = [&](const unsigned short* lds, int tr) -> v16bf {
        Frag f;
        const unsigned short* p = lds + (wr * 32 + tr * 16 + l16) * LBS + half * 8;
        f.q[0] = *(const uint4*)(p);
        f.q[1] = *(const uint4*)(p + 16);
        return f.v;
    };
    // ISA 16-bit B 32x16: lane = col (lane%16), holds K[8*half..+8) in v0-3, K[16+8*half..+8) in v4-7.
    auto loadB = [&](const unsigned short* lds, int tc) -> v16bf {
        Frag f;
        const unsigned short* p = lds + (wc * 64 + tc * 16 + l16) * LBS + half * 8;
        f.q[0] = *(const uint4*)(p);
        f.q[1] = *(const uint4*)(p + 16);
        return f.v;
    };

    v8f acc[2][4];
    #pragma unroll
    for (int tr = 0; tr < 2; ++tr)
        #pragma unroll
        for (int tc = 0; tc < 4; ++tc)
            acc[tr][tc] = (v8f)0.0f;

    for (int kc = 0; kc < DIM; kc += 32) {
        __syncthreads();
        // Copy phase: stage all 8 quads in distinct temps so the 8
        // global_load_b128 issue back-to-back and drain with partial
        // s_wait_loadcnt.  Live range is the copy phase only -> no spill.
        {
            const uint4 t0 = *(const uint4*)(Ehi + rowA0 + kc);
            const uint4 t1 = *(const uint4*)(Ehi + rowA1 + kc);
            const uint4 t2 = *(const uint4*)(Elo + rowA0 + kc);
            const uint4 t3 = *(const uint4*)(Elo + rowA1 + kc);
            const uint4 t4 = *(const uint4*)(Ehi + rowB0 + kc);
            const uint4 t5 = *(const uint4*)(Ehi + rowB1 + kc);
            const uint4 t6 = *(const uint4*)(Elo + rowB0 + kc);
            const uint4 t7 = *(const uint4*)(Elo + rowB1 + kc);
            *(uint4*)(ldsAhi + ld0) = t0;
            *(uint4*)(ldsAhi + ld1) = t1;
            *(uint4*)(ldsAlo + ld0) = t2;
            *(uint4*)(ldsAlo + ld1) = t3;
            *(uint4*)(ldsBhi + ld0) = t4;
            *(uint4*)(ldsBhi + ld1) = t5;
            *(uint4*)(ldsBlo + ld0) = t6;
            *(uint4*)(ldsBlo + ld1) = t7;
        }
        __syncthreads();

        if (kc + 32 < DIM) {
            // Pull next chunk's lines toward the WGP (no VGPR/counter cost).
            __builtin_prefetch(Ehi + rowA0 + kc + 32, 0, 3);
            __builtin_prefetch(Elo + rowA0 + kc + 32, 0, 3);
            __builtin_prefetch(Ehi + rowB0 + kc + 32, 0, 3);
            __builtin_prefetch(Elo + rowB0 + kc + 32, 0, 3);
        }

        v16bf ah[2], al[2];
        #pragma unroll
        for (int tr = 0; tr < 2; ++tr) {
            ah[tr] = loadA(ldsAhi, tr);
            al[tr] = loadA(ldsAlo, tr);
        }
        #pragma unroll
        for (int tc = 0; tc < 4; ++tc) {
            const v16bf bh = loadB(ldsBhi, tc);
            const v16bf bl = loadB(ldsBlo, tc);
            #pragma unroll
            for (int tr = 0; tr < 2; ++tr) {
                acc[tr][tc] = __builtin_amdgcn_wmma_f32_16x16x32_bf16(
                    false, ah[tr], false, bh, (short)0, acc[tr][tc], false, false);
                acc[tr][tc] = __builtin_amdgcn_wmma_f32_16x16x32_bf16(
                    false, ah[tr], false, bl, (short)0, acc[tr][tc], false, false);
                acc[tr][tc] = __builtin_amdgcn_wmma_f32_16x16x32_bf16(
                    false, al[tr], false, bh, (short)0, acc[tr][tc], false, false);
            }
        }
    }

    float s = 0.0f;
    int   cnt = 0;
    #pragma unroll
    for (int tr = 0; tr < 2; ++tr) {
        #pragma unroll
        for (int tc = 0; tc < 4; ++tc) {
            const int lj   = wc * 64 + tc * 16 + l16;
            const int gj   = bx * BT + lj;
            const int labj = slB[lj];
            const float x2j = sx2B[lj];
            #pragma unroll
            for (int r = 0; r < 8; ++r) {
                const int lrow = wr * 32 + tr * 16 + r + 8 * half;
                const int gi   = by * BT + lrow;
                if (gi < gj && slA[lrow] == labj) {
                    float sq = sx2A[lrow] + x2j - 2.0f * acc[tr][tc][r];
                    sq = fmaxf(sq, 0.0f);
                    s += sqrtf(sq);
                    ++cnt;
                }
            }
        }
    }

    #pragma unroll
    for (int off = 16; off > 0; off >>= 1) {
        s   += __shfl_down(s, off, 32);
        cnt += __shfl_down(cnt, off, 32);
    }
    if (lane == 0) { red_s[wave] = s; red_c[wave] = cnt; }
    __syncthreads();
    if (tid == 0) {
        float S = 0.0f; int C = 0;
        #pragma unroll
        for (int w = 0; w < 8; ++w) { S += red_s[w]; C += red_c[w]; }
        atomicAdd(&accum[0], S);
        atomicAdd(&accum[1], (float)C);
    }
}

// ---------------------------------------------------------------------------
// Fallback path: native fp32 WMMA (V_WMMA_F32_16X16X4_F32), used only if the
// workspace can't hold the bf16 split copies.
// ---------------------------------------------------------------------------
__global__ __launch_bounds__(256) void oc_pairdist_f32_kernel(const float* __restrict__ E,
                                                              const int*   __restrict__ labels,
                                                              const float* __restrict__ x2,
                                                              float*       __restrict__ accum) {
    const int bx = blockIdx.x;
    const int by = blockIdx.y;
    if (bx < by) return;

    __shared__ __align__(16) float ldsA[BT * LDSS];
    __shared__ __align__(16) float ldsB[BT * LDSS];
    __shared__ float sx2A[BT], sx2B[BT];
    __shared__ int   slA[BT],  slB[BT];
    __shared__ float red_s[8];
    __shared__ int   red_c[8];

    const int tid  = threadIdx.x;
    const int wave = tid >> 5;
    const int lane = tid & 31;
    const int half = lane >> 4;
    const int l16  = lane & 15;
    const int wr   = wave >> 1;
    const int wc   = wave & 1;

    if (tid < BT) {
        sx2A[tid] = x2[by * BT + tid];
        slA[tid]  = labels[by * BT + tid];
    } else {
        const int t = tid - BT;
        sx2B[t] = x2[bx * BT + t];
        slB[t]  = labels[bx * BT + t];
    }

    v8f acc[2][4];
    #pragma unroll
    for (int tr = 0; tr < 2; ++tr)
        #pragma unroll
        for (int tc = 0; tc < 4; ++tc)
            acc[tr][tc] = (v8f)0.0f;

    for (int kc = 0; kc < DIM; kc += KC) {
        __syncthreads();
        {
            // Stage 8 float4 in distinct temps (same batching as bf16 path).
            float4 ta[4], tb[4];
            #pragma unroll
            for (int t = 0; t < 4; ++t) {
                const int fi  = tid + t * 256;
                const int row = fi >> 3;
                const int c4  = (fi & 7) << 2;
                ta[t] = *(const float4*)(E + (size_t)(by * BT + row) * DIM + kc + c4);
                tb[t] = *(const float4*)(E + (size_t)(bx * BT + row) * DIM + kc + c4);
            }
            #pragma unroll
            for (int t = 0; t < 4; ++t) {
                const int fi  = tid + t * 256;
                const int row = fi >> 3;
                const int c4  = (fi & 7) << 2;
                *(float4*)(ldsA + row * LDSS + c4) = ta[t];
                *(float4*)(ldsB + row * LDSS + c4) = tb[t];
            }
        }
        __syncthreads();

        #pragma unroll
        for (int ks = 0; ks < KC; ks += 4) {
            const int col = ks + 2 * half;
            v2f afrag[2], bfrag[4];
            #pragma unroll
            for (int tr = 0; tr < 2; ++tr)
                afrag[tr] = *(const v2f*)(ldsA + (wr * 32 + tr * 16 + l16) * LDSS + col);
            #pragma unroll
            for (int tc = 0; tc < 4; ++tc)
                bfrag[tc] = *(const v2f*)(ldsB + (wc * 64 + tc * 16 + l16) * LDSS + col);
            #pragma unroll
            for (int tr = 0; tr < 2; ++tr)
                #pragma unroll
                for (int tc = 0; tc < 4; ++tc)
                    acc[tr][tc] = __builtin_amdgcn_wmma_f32_16x16x4_f32(
                        false, afrag[tr], false, bfrag[tc],
                        (short)0, acc[tr][tc], false, false);
        }
    }

    float s = 0.0f;
    int   cnt = 0;
    #pragma unroll
    for (int tr = 0; tr < 2; ++tr) {
        #pragma unroll
        for (int tc = 0; tc < 4; ++tc) {
            const int lj   = wc * 64 + tc * 16 + l16;
            const int gj   = bx * BT + lj;
            const int labj = slB[lj];
            const float x2j = sx2B[lj];
            #pragma unroll
            for (int r = 0; r < 8; ++r) {
                const int lrow = wr * 32 + tr * 16 + r + 8 * half;
                const int gi   = by * BT + lrow;
                if (gi < gj && slA[lrow] == labj) {
                    float sq = sx2A[lrow] + x2j - 2.0f * acc[tr][tc][r];
                    sq = fmaxf(sq, 0.0f);
                    s += sqrtf(sq);
                    ++cnt;
                }
            }
        }
    }

    #pragma unroll
    for (int off = 16; off > 0; off >>= 1) {
        s   += __shfl_down(s, off, 32);
        cnt += __shfl_down(cnt, off, 32);
    }
    if (lane == 0) { red_s[wave] = s; red_c[wave] = cnt; }
    __syncthreads();
    if (tid == 0) {
        float S = 0.0f; int C = 0;
        #pragma unroll
        for (int w = 0; w < 8; ++w) { S += red_s[w]; C += red_c[w]; }
        atomicAdd(&accum[0], S);
        atomicAdd(&accum[1], (float)C);
    }
}

__global__ void oc_final_kernel(const float* ws, float* out) {
    out[0] = ws[0] / ws[1];
}

extern "C" void kernel_launch(void* const* d_in, const int* in_sizes, int n_in,
                              void* d_out, int out_size, void* d_ws, size_t ws_size,
                              hipStream_t stream) {
    const float* E      = (const float*)d_in[0];   // [8192,256] fp32
    const int*   labels = (const int*)d_in[1];     // [8192] int32
    float*       out    = (float*)d_out;           // scalar fp32
    float*       ws     = (float*)d_ws;

    (void)in_sizes; (void)n_in; (void)out_size;

    const size_t HI_OFF   = 16 + (size_t)NROW * 4;                 // 32784 B, 16B-aligned
    const size_t LO_OFF   = HI_OFF + (size_t)NROW * DIM * 2;       // +4 MB
    const size_t NEED_B16 = LO_OFF + (size_t)NROW * DIM * 2;       // ~8.03 MB

    float* x2p = ws + 4;   // byte offset 16

    oc_init_kernel<<<1, 1, 0, stream>>>(ws);
    oc_rowsq_kernel<<<NROW / 8, 256, 0, stream>>>(E, x2p);

    dim3 grid(NROW / BT, NROW / BT);
    if (ws_size >= NEED_B16) {
        unsigned short* hi = (unsigned short*)((char*)d_ws + HI_OFF);
        unsigned short* lo = (unsigned short*)((char*)d_ws + LO_OFF);
        oc_split_kernel<<<(NROW * DIM / 4) / 256, 256, 0, stream>>>(E, hi, lo);
        oc_pairdist_bf16_kernel<<<grid, 256, 0, stream>>>(hi, lo, labels, x2p, ws);
    } else {
        oc_pairdist_f32_kernel<<<grid, 256, 0, stream>>>(E, labels, x2p, ws);
    }
    oc_final_kernel<<<1, 1, 0, stream>>>(ws, out);
}
